// Net_62191126446131
// MI455X (gfx1250) — matrix operand order repaired and checked by reading, not compiled, not run on hardware
//
#include <hip/hip_runtime.h>
#include <hip/hip_bf16.h>
#include <math.h>

// ---------------- model constants ----------------
static constexpr int Bc = 4, Lc = 512, IN_DIM = 32, OUT_DIM = 1;
static constexpr int D_MODEL = 768, N_LAYERS = 4;
static constexpr int E = 2 * D_MODEL;        // 1536
static constexpr int TWO_E = 2 * E;          // 3072
static constexpr int Nst = 16, Kc = 4, R = D_MODEL / 16;  // 48
static constexpr int RPAD = 64;              // dt padded K (multiple of 32)
static constexpr int RN2 = R + 2 * Nst;      // 80
static constexpr int M = Bc * Lc;            // 2048 flattened rows

typedef __attribute__((ext_vector_type(16))) _Float16 v16h;
typedef __attribute__((ext_vector_type(8)))  _Float16 v8h;
typedef __attribute__((ext_vector_type(8)))  float    v8f;

// ---------------- fp32 -> fp16 convert ----------------
__global__ void k_f32_to_f16(const float* __restrict__ src, _Float16* __restrict__ dst, int n) {
    int i = blockIdx.x * 256 + threadIdx.x;
    if (i < n) dst[i] = (_Float16)src[i];
}

// ---------------- rmsnorm -> f16 ----------------
__global__ void k_rmsnorm_f16(const float* __restrict__ x, const float* __restrict__ w,
                              _Float16* __restrict__ out, int ncols) {
    __shared__ float red[256];
    const int row = blockIdx.x, tid = threadIdx.x;
    const float* xr = x + (size_t)row * ncols;
    float s = 0.f;
    for (int c = tid; c < ncols; c += 256) { float v = xr[c]; s += v * v; }
    red[tid] = s; __syncthreads();
    for (int off = 128; off > 0; off >>= 1) {
        if (tid < off) red[tid] += red[tid + off];
        __syncthreads();
    }
    const float rinv = rsqrtf(red[0] / (float)ncols + 1e-5f);
    for (int c = tid; c < ncols; c += 256)
        out[(size_t)row * ncols + c] = (_Float16)(xr[c] * rinv * w[c]);
}

// ---------------- WMMA GEMM:  C[M x N] (+)= A[M x K] * W[N x K]^T ----------------
// Requirements: K % 32 == 0, M % 128 == 0. N-tail handled by clamped (in-bounds)
// reads + guarded stores. mode: 0=store, 1=+bias, 2=softplus(v+bias), 3=C+=v.
__device__ __forceinline__ v16h load_afrag(const _Float16* p) {
    // lane already offset by lhalf*8; halfs [0..7] and [16..23] relative to p.
    v8h lo = *(const v8h*)p;
    v8h hi = *(const v8h*)(p + 16);
    v16h af;
#pragma unroll
    for (int i = 0; i < 8; ++i) { af[i] = lo[i]; af[i + 8] = hi[i]; }
    return af;
}

__global__ void __launch_bounds__(128)
k_gemm_wmma(const _Float16* __restrict__ A, const _Float16* __restrict__ Bw,
            const float* __restrict__ bias, float* __restrict__ C,
            int Mr, int Nc, int Kd, int mode) {
    const int lane  = threadIdx.x & 31;
    const int wave  = threadIdx.x >> 5;
    const int lrow  = lane & 15;
    const int lhalf = lane >> 4;
    const int m0 = blockIdx.y * 128 + wave * 32;   // 2 x 16 rows per wave
    const int n0 = blockIdx.x * 64;                // 4 x 16 cols per wave

    // loop-invariant, in-bounds B row pointers (clamped for the N tail)
    const _Float16* brow[4];
#pragma unroll
    for (int t = 0; t < 4; ++t) {
        int n = n0 + t * 16 + lrow;
        if (n >= Nc) n = Nc - 1;
        brow[t] = Bw + (size_t)n * Kd + lhalf * 8;
    }
    const _Float16* arow0 = A + (size_t)(m0 + lrow) * Kd + lhalf * 8;
    const _Float16* arow1 = arow0 + (size_t)16 * Kd;

    v8f acc[2][4] = {};
    for (int k0 = 0; k0 < Kd; k0 += 32) {
        const v16h a0 = load_afrag(arow0 + k0);
        const v16h a1 = load_afrag(arow1 + k0);
        v16h bf[4];
#pragma unroll
        for (int t = 0; t < 4; ++t) bf[t] = load_afrag(brow[t] + k0);
#pragma unroll
        for (int t = 0; t < 4; ++t)
            acc[0][t] = __builtin_amdgcn_wmma_f32_16x16x32_f16(false, a0, false, bf[t],
                                                               (short)0, acc[0][t], false, false);
#pragma unroll
        for (int t = 0; t < 4; ++t)
            acc[1][t] = __builtin_amdgcn_wmma_f32_16x16x32_f16(false, a1, false, bf[t],
                                                               (short)0, acc[1][t], false, false);
        if (k0 + 32 < Kd) {  // uniform scalar branch, no exec churn
            __builtin_prefetch(arow0 + k0 + 32, 0, 3);  // global_prefetch_b8
            __builtin_prefetch(arow1 + k0 + 32, 0, 3);
        }
    }

    // epilogue: lane covers col n0+t*16+lrow, rows m0 + h*16 + lhalf*8 + r
#pragma unroll
    for (int h = 0; h < 2; ++h) {
#pragma unroll
        for (int t = 0; t < 4; ++t) {
            const int n = n0 + t * 16 + lrow;
            if (n >= Nc) continue;
            const float bv = (mode == 1 || mode == 2) ? bias[n] : 0.f;
#pragma unroll
            for (int r = 0; r < 8; ++r) {
                const int m = m0 + h * 16 + lhalf * 8 + r;
                if (m >= Mr) continue;
                float v = acc[h][t][r] + bv;
                if (mode == 2) v = (v > 20.f) ? v : log1pf(__expf(v));
                const size_t ci = (size_t)m * Nc + n;
                if (mode == 3) C[ci] += v; else C[ci] = v;
            }
        }
    }
}

// ---------------- causal depthwise conv (K=4) + bias + silu ----------------
__global__ void k_conv_silu(const float* __restrict__ xz, const float* __restrict__ cw,
                            const float* __restrict__ cb, float* __restrict__ u,
                            _Float16* __restrict__ u_h) {
    const int idx = blockIdx.x * 256 + threadIdx.x;
    if (idx >= M * E) return;
    const int e = idx % E;
    const int bl = idx / E;
    const int l = bl % Lc, b = bl / Lc;
    float acc = cb[e];
    const float* base = xz + (size_t)(b * Lc) * TWO_E + e;  // xc = xz[..., :E]
#pragma unroll
    for (int j = 0; j < Kc; ++j) {
        const int t = l - (Kc - 1) + j;
        if (t >= 0) acc += cw[e * Kc + j] * base[(size_t)t * TWO_E];
    }
    const float sv = acc / (1.f + __expf(-acc));  // silu
    u[idx] = sv;
    u_h[idx] = (_Float16)sv;
}

// ---------------- extract dt (first R cols of x_dbl), zero-padded to RPAD ----------------
__global__ void k_extract_dt(const float* __restrict__ x_dbl, _Float16* __restrict__ dt_h) {
    const int idx = blockIdx.x * 256 + threadIdx.x;
    if (idx >= M * RPAD) return;
    const int m = idx / RPAD, r = idx % RPAD;
    dt_h[idx] = (r < R) ? (_Float16)x_dbl[(size_t)m * RN2 + r] : (_Float16)0.f;
}

// ---------------- selective scan (sequential over L, parallel over B x E) ----------------
__global__ void __launch_bounds__(256)
k_scan(const float* __restrict__ delta, const float* __restrict__ u,
       const float* __restrict__ x_dbl, const float* __restrict__ A_log,
       const float* __restrict__ Dssm, float* __restrict__ ys) {
    const int b = blockIdx.x;
    const int e = blockIdx.y * 256 + threadIdx.x;
    __shared__ float BmC[32];  // [0..15]=Bm(t), [16..31]=C(t)
    float Arow[Nst];
#pragma unroll
    for (int n = 0; n < Nst; ++n) Arow[n] = -__expf(A_log[(size_t)e * Nst + n]);
    const float Dv = Dssm[e];
    float h[Nst];
#pragma unroll
    for (int n = 0; n < Nst; ++n) h[n] = 0.f;

    for (int t = 0; t < Lc; ++t) {
        const size_t row = (size_t)(b * Lc + t);
        if (threadIdx.x < 32) BmC[threadIdx.x] = x_dbl[row * RN2 + R + threadIdx.x];
        __syncthreads();
        const float dv = delta[row * E + e];
        const float uv = u[row * E + e];
        const float dBu = dv * uv;
        float y = 0.f;
#pragma unroll
        for (int n = 0; n < Nst; ++n) {
            h[n] = __expf(dv * Arow[n]) * h[n] + dBu * BmC[n];
            y += h[n] * BmC[16 + n];
        }
        ys[row * E + e] = y + uv * Dv;
        __syncthreads();
    }
}

// ---------------- gate: yg = ys * silu(res), as f16 ----------------
__global__ void k_gate(const float* __restrict__ ys, const float* __restrict__ xz,
                       _Float16* __restrict__ yg_h) {
    const int idx = blockIdx.x * 256 + threadIdx.x;
    if (idx >= M * E) return;
    const int e = idx % E, m = idx / E;
    const float res = xz[(size_t)m * TWO_E + E + e];
    const float g = res / (1.f + __expf(-res));
    yg_h[idx] = (_Float16)(ys[idx] * g);
}

// ---------------- final: rmsnorm(last token) . out_w + out_b, sigmoid ----------------
__global__ void k_final(const float* __restrict__ h, const float* __restrict__ normf_w,
                        const float* __restrict__ out_w, const float* __restrict__ out_b,
                        float* __restrict__ out) {
    __shared__ float red[256];
    const int b = blockIdx.x, tid = threadIdx.x;
    const float* row = h + (size_t)(b * Lc + (Lc - 1)) * D_MODEL;
    float s = 0.f;
    for (int c = tid; c < D_MODEL; c += 256) { float v = row[c]; s += v * v; }
    red[tid] = s; __syncthreads();
    for (int off = 128; off > 0; off >>= 1) {
        if (tid < off) red[tid] += red[tid + off];
        __syncthreads();
    }
    const float rinv = rsqrtf(red[0] / (float)D_MODEL + 1e-5f);
    __syncthreads();
    float d = 0.f;
    for (int c = tid; c < D_MODEL; c += 256) d += row[c] * rinv * normf_w[c] * out_w[c];
    red[tid] = d; __syncthreads();
    for (int off = 128; off > 0; off >>= 1) {
        if (tid < off) red[tid] += red[tid + off];
        __syncthreads();
    }
    if (tid == 0) {
        const float z = red[0] + out_b[0];
        out[b] = 1.f / (1.f + __expf(-z));
    }
}

// ---------------- host side ----------------
static void launch_gemm(const _Float16* A, const _Float16* W, const float* bias, float* C,
                        int m, int n, int k, int mode, hipStream_t s) {
    dim3 grid((n + 63) / 64, (m + 127) / 128);
    k_gemm_wmma<<<grid, 128, 0, s>>>(A, W, bias, C, m, n, k, mode);
}
static void launch_conv(const float* src, _Float16* dst, int n, hipStream_t s) {
    k_f32_to_f16<<<(n + 255) / 256, 256, 0, s>>>(src, dst, n);
}

extern "C" void kernel_launch(void* const* d_in, const int* in_sizes, int n_in,
                              void* d_out, int out_size, void* d_ws, size_t ws_size,
                              hipStream_t stream) {
    (void)in_sizes; (void)n_in; (void)out_size; (void)ws_size;
    const float* x         = (const float*)d_in[0];
    const float* in_w      = (const float*)d_in[1];
    const float* in_b      = (const float*)d_in[2];
    const float* in_proj_w = (const float*)d_in[3];
    const float* conv_w    = (const float*)d_in[4];
    const float* conv_b    = (const float*)d_in[5];
    const float* xproj_w   = (const float*)d_in[6];
    const float* dtproj_w  = (const float*)d_in[7];
    const float* dtproj_b  = (const float*)d_in[8];
    const float* A_log     = (const float*)d_in[9];
    const float* D_ssm     = (const float*)d_in[10];
    const float* outproj_w = (const float*)d_in[11];
    const float* norm_w    = (const float*)d_in[12];
    const float* normf_w   = (const float*)d_in[13];
    const float* out_w     = (const float*)d_in[14];
    const float* out_b     = (const float*)d_in[15];

    char* base = (char*)d_ws;
    size_t off = 0;
    auto carve = [&](size_t bytes) -> void* {
        void* p = base + off;
        off += (bytes + 255) & ~(size_t)255;
        return p;
    };

    _Float16* x_h       = (_Float16*)carve((size_t)M * IN_DIM * 2);
    _Float16* in_w_h    = (_Float16*)carve((size_t)D_MODEL * IN_DIM * 2);
    float*    hbuf      = (float*)   carve((size_t)M * D_MODEL * 4);
    _Float16* xn_h      = (_Float16*)carve((size_t)M * D_MODEL * 2);
    _Float16* in_proj_h = (_Float16*)carve((size_t)N_LAYERS * TWO_E * D_MODEL * 2);
    float*    xz        = (float*)   carve((size_t)M * TWO_E * 4);
    float*    u         = (float*)   carve((size_t)M * E * 4);
    _Float16* u_h       = (_Float16*)carve((size_t)M * E * 2);
    _Float16* xproj_h   = (_Float16*)carve((size_t)N_LAYERS * RN2 * E * 2);
    float*    x_dbl     = (float*)   carve((size_t)M * RN2 * 4);
    _Float16* dt_h      = (_Float16*)carve((size_t)M * RPAD * 2);
    _Float16* dtproj_h  = (_Float16*)carve((size_t)N_LAYERS * E * R * 2);
    float*    delta     = (float*)   carve((size_t)M * E * 4);
    float*    ys        = (float*)   carve((size_t)M * E * 4);
    _Float16* yg_h      = (_Float16*)carve((size_t)M * E * 2);
    _Float16* outproj_h = (_Float16*)carve((size_t)N_LAYERS * D_MODEL * E * 2);
    // dtproj padded to K=RPAD (zero cols R..RPAD-1), per layer
    _Float16* dtproj_pad = (_Float16*)carve((size_t)N_LAYERS * E * RPAD * 2);

    // ---- conversions to f16 (redone every call; deterministic) ----
    launch_conv(x,         x_h,       M * IN_DIM, stream);
    launch_conv(in_w,      in_w_h,    D_MODEL * IN_DIM, stream);
    launch_conv(in_proj_w, in_proj_h, N_LAYERS * TWO_E * D_MODEL, stream);
    launch_conv(xproj_w,   xproj_h,   N_LAYERS * RN2 * E, stream);
    launch_conv(dtproj_w,  dtproj_h,  N_LAYERS * E * R, stream);
    launch_conv(outproj_w, outproj_h, N_LAYERS * D_MODEL * E, stream);
    // pad dtproj rows R->RPAD with zeros (reuse extract-style kernel via lambda kernel)
    {
        // k_extract_dt pads activations; for the weight we pad with a dedicated pass:
        // dt_pad[e, r] = (r < R) ? dtproj_h[e, r] : 0
        // implemented with k_pad below
        extern __global__ void k_pad_dtw(const _Float16*, _Float16*, int);
    }
    // (see k_pad_dtw definition below main; launched here)
    {
        void* args_unused = nullptr; (void)args_unused;
    }
    // pad weights
    {
        const int n = N_LAYERS * E * RPAD;
        struct Launch { };
        (void)sizeof(Launch);
        // actual launch below via function
    }
    // --- real pad launch ---
    {
        extern void launch_pad_dtw(const _Float16*, _Float16*, hipStream_t);
        launch_pad_dtw(dtproj_h, dtproj_pad, stream);
    }

    // ---- input projection: h = x @ in_w^T + in_b ----
    launch_gemm(x_h, in_w_h, in_b, hbuf, M, D_MODEL, IN_DIM, /*mode=*/1, stream);

    for (int l = 0; l < N_LAYERS; ++l) {
        const _Float16* ipw = in_proj_h  + (size_t)l * TWO_E * D_MODEL;
        const _Float16* xpw = xproj_h    + (size_t)l * RN2 * E;
        const _Float16* dpw = dtproj_pad + (size_t)l * E * RPAD;
        const _Float16* opw = outproj_h  + (size_t)l * D_MODEL * E;

        // rmsnorm -> f16
        k_rmsnorm_f16<<<M, 256, 0, stream>>>(hbuf, norm_w + (size_t)l * D_MODEL, xn_h, D_MODEL);
        // xz = xn @ in_proj^T
        launch_gemm(xn_h, ipw, nullptr, xz, M, TWO_E, D_MODEL, 0, stream);
        // depthwise causal conv + bias + silu -> u, u_h
        k_conv_silu<<<(M * E + 255) / 256, 256, 0, stream>>>(
            xz, conv_w + (size_t)l * E * Kc, conv_b + (size_t)l * E, u, u_h);
        // x_dbl = u @ xproj^T   (dt | Bm | C)
        launch_gemm(u_h, xpw, nullptr, x_dbl, M, RN2, E, 0, stream);
        // dt slice -> f16, zero-padded K=RPAD
        k_extract_dt<<<(M * RPAD + 255) / 256, 256, 0, stream>>>(x_dbl, dt_h);
        // delta = softplus(dt @ dtproj^T + dtproj_b), K = RPAD (padded, mult of 32)
        launch_gemm(dt_h, dpw, dtproj_b + (size_t)l * E, delta, M, E, RPAD, /*mode=*/2, stream);
        // selective scan
        dim3 sg(Bc, E / 256);
        k_scan<<<sg, 256, 0, stream>>>(delta, u, x_dbl,
                                       A_log + (size_t)l * E * Nst,
                                       D_ssm + (size_t)l * E, ys);
        // gate with silu(res)
        k_gate<<<(M * E + 255) / 256, 256, 0, stream>>>(ys, xz, yg_h);
        // residual: h += yg @ outproj^T
        launch_gemm(yg_h, opw, nullptr, hbuf, M, D_MODEL, E, /*mode=*/3, stream);
    }

    // final rmsnorm on last token + linear + sigmoid
    k_final<<<Bc, 256, 0, stream>>>(hbuf, normf_w, out_w, out_b, (float*)d_out);
}

// ---- dtproj zero-pad helper (K=R -> K=RPAD) ----
__global__ void k_pad_dtw(const _Float16* __restrict__ src, _Float16* __restrict__ dst, int n) {
    const int idx = blockIdx.x * 256 + threadIdx.x;
    if (idx >= n) return;
    const int r = idx % RPAD, e = idx / RPAD;
    dst[idx] = (r < R) ? src[(size_t)e * R + r] : (_Float16)0.f;
}
void launch_pad_dtw(const _Float16* src, _Float16* dst, hipStream_t s) {
    const int n = N_LAYERS * E * RPAD;
    k_pad_dtw<<<(n + 255) / 256, 256, 0, s>>>(src, dst, n);
}
